// UserTransformerEncoderLayer_24215025614965
// MI455X (gfx1250) — matrix operand order repaired and checked by reading, not compiled, not run on hardware
//
#include <hip/hip_runtime.h>
#include <hip/hip_bf16.h>
#include <cmath>

// ---------------------------------------------------------------------------
// CDNA5 (gfx1250) transformer encoder layer: bf16 WMMA everywhere.
// ---------------------------------------------------------------------------

typedef __bf16 bf16;
typedef __attribute__((ext_vector_type(16))) __bf16 v16bf;
typedef __attribute__((ext_vector_type(8)))  float  v8f;

#define NSEQ   8192
#define DMODEL 256
#define DHID   1024
#define ATTN_SCALE 0.0625f   // 1/sqrt(256)
#define LN_EPS 1e-5f

// ---- WMMA fragment loaders (wave32, 16x16x32 bf16) ------------------------
// A operand: 16x32 (MxK), row-major source with leading dim ld.
// Layout: lanes 0-15 -> row=lane, K 0..15; lanes 16-31 -> row=lane-16, K 16..31
__device__ __forceinline__ v16bf load_a_tile(const bf16* __restrict__ src, int ld) {
  const int lane = threadIdx.x & 31;
  const int row  = lane & 15;
  const int koff = (lane >> 4) << 4;          // 0 or 16
  const bf16* p = src + (size_t)row * ld;
  v16bf a;
#pragma unroll
  for (int i = 0; i < 8; ++i) a[i]     = p[koff + i];
#pragma unroll
  for (int i = 0; i < 8; ++i) a[8 + i] = p[koff + 8 + i];
  return a;
}

// B operand: 32x16 (KxN). Source is "B^T" row-major: row n holds K contiguous.
// Layout: lanes 0-15 -> col=lane, K 0..15; lanes 16-31 -> col=lane-16, K 16..31
__device__ __forceinline__ v16bf load_b_tile(const bf16* __restrict__ srcT, int ld) {
  const int lane  = threadIdx.x & 31;
  const int col   = lane & 15;
  const int kbase = (lane >> 4) << 4;         // 0 or 16
  const bf16* p = srcT + (size_t)col * ld + kbase;
  v16bf b;
#pragma unroll
  for (int i = 0; i < 16; ++i) b[i] = p[i];
  return b;
}

__device__ __forceinline__ v8f wmma_bf16(v16bf a, v16bf b, v8f c) {
  return __builtin_amdgcn_wmma_f32_16x16x32_bf16(false, a, false, b, (short)0, c,
                                                 false, false);
}

// ---- prep kernels ----------------------------------------------------------
__global__ void cvt_bf16_kernel(const float* __restrict__ in, bf16* __restrict__ out, int n) {
  int i = blockIdx.x * 256 + threadIdx.x;
  if (i < n) out[i] = (bf16)in[i];
}

// in [R,C] f32 -> out [C,R] bf16
__global__ void transpose_cvt_kernel(const float* __restrict__ in, bf16* __restrict__ out,
                                     int R, int C) {
  int i = blockIdx.x * 256 + threadIdx.x;
  if (i < R * C) {
    int r = i / C, c = i % C;
    out[(size_t)c * R + r] = (bf16)in[i];
  }
}

// ---- generic GEMM + bias (+relu), bf16 out, optional transposed store ------
// A: [M,K] bf16 row-major.  Bt: [Nout,K] bf16 (pre-transposed weights).
__global__ void wmma_gemm_bias_kernel(const bf16* __restrict__ A,
                                      const bf16* __restrict__ Bt,
                                      const float* __restrict__ bias,
                                      bf16* __restrict__ out,
                                      int M, int Nout, int K,
                                      int relu, int store_t) {
  const int wave = threadIdx.x >> 5;
  const int lane = threadIdx.x & 31;
  const int ntN  = Nout >> 4;
  const int tile = blockIdx.x * 8 + wave;
  if (tile >= (M >> 4) * ntN) return;
  const int mi = tile / ntN, ni = tile % ntN;
  const bf16* Ab = A  + (size_t)mi * 16 * K;
  const bf16* Bb = Bt + (size_t)ni * 16 * K;
  v8f acc = {};
  for (int kt = 0; kt < (K >> 5); ++kt) {
    v16bf a = load_a_tile(Ab + kt * 32, K);
    v16bf b = load_b_tile(Bb + kt * 32, K);
    acc = wmma_bf16(a, b, acc);
  }
  const int col = lane & 15;
  const int rb  = (lane >> 4) << 3;
  const float bv = bias[ni * 16 + col];
#pragma unroll
  for (int i = 0; i < 8; ++i) {
    float v = acc[i] + bv;
    if (relu) v = fmaxf(v, 0.f);
    const int r = mi * 16 + rb + i;
    const int c = ni * 16 + col;
    if (store_t) out[(size_t)c * M + r]    = (bf16)v;
    else         out[(size_t)r * Nout + c] = (bf16)v;
  }
}

// ---- pass A: per-row softmax stats (max m, sum l) over S = Qp Kp^T * s + mask
__global__ void softmax_stats_kernel(const bf16* __restrict__ qp,
                                     const bf16* __restrict__ kp,
                                     const float* __restrict__ mask,
                                     float* __restrict__ Mrow,
                                     float* __restrict__ Lrow) {
  const int wave = threadIdx.x >> 5, lane = threadIdx.x & 31;
  const int rowbase = blockIdx.x * 128 + wave * 16;
  const int col = lane & 15, rb = (lane >> 4) << 3;

  v16bf qa[8];
#pragma unroll
  for (int kt = 0; kt < 8; ++kt)
    qa[kt] = load_a_tile(qp + (size_t)rowbase * DMODEL + kt * 32, DMODEL);

  float m[8], l[8];
#pragma unroll
  for (int i = 0; i < 8; ++i) { m[i] = -INFINITY; l[i] = 0.f; }

  for (int jt = 0; jt < NSEQ / 16; ++jt) {
    v8f s = {};
#pragma unroll
    for (int kt = 0; kt < 8; ++kt) {
      v16bf b = load_b_tile(kp + (size_t)jt * 16 * DMODEL + kt * 32, DMODEL);
      s = wmma_bf16(qa[kt], b, s);
    }
    const int c = jt * 16 + col;
    if (jt + 1 < NSEQ / 16)  // stream hint on the dominant mask traffic
      __builtin_prefetch(&mask[(size_t)(rowbase + rb) * NSEQ + c + 16], 0, 0);
#pragma unroll
    for (int i = 0; i < 8; ++i) {
      float v  = s[i] * ATTN_SCALE + mask[(size_t)(rowbase + rb + i) * NSEQ + c];
      float nm = fmaxf(m[i], v);
      l[i] = l[i] * __expf(m[i] - nm) + __expf(v - nm);
      m[i] = nm;
    }
  }
  // merge the 16 per-lane column partitions (stays within each 16-lane half)
#pragma unroll
  for (int d = 1; d < 16; d <<= 1) {
#pragma unroll
    for (int i = 0; i < 8; ++i) {
      float om = __shfl_xor(m[i], d, 32);
      float ol = __shfl_xor(l[i], d, 32);
      float nm = fmaxf(m[i], om);
      l[i] = l[i] * __expf(m[i] - nm) + ol * __expf(om - nm);
      m[i] = nm;
    }
  }
  if (col == 0) {
#pragma unroll
    for (int i = 0; i < 8; ++i) {
      Mrow[rowbase + rb + i] = m[i];
      Lrow[rowbase + rb + i] = l[i];
    }
  }
}

// ---- pass B: P = exp(S-m)/l + pref, out = P @ Vp  (Vp pre-transposed) ------
// Workgroup: 16 query rows. Stage 1: 8 waves compute a [16,128] P chunk -> LDS.
// Stage 2: each wave accumulates 2 of the 16 output d-column tiles.
__global__ void attn_pv_kernel(const bf16* __restrict__ qp,
                               const bf16* __restrict__ kp,
                               const bf16* __restrict__ vpT,
                               const float* __restrict__ mask,
                               const float* __restrict__ pref,
                               const float* __restrict__ Mrow,
                               const float* __restrict__ Lrow,
                               bf16* __restrict__ avb) {
  __shared__ bf16 ldsP[16 * 128];
  const int wave = threadIdx.x >> 5, lane = threadIdx.x & 31;
  const int rowbase = blockIdx.x * 16;
  const int col = lane & 15, rb = (lane >> 4) << 3;

  v16bf qa[8];
#pragma unroll
  for (int kt = 0; kt < 8; ++kt)
    qa[kt] = load_a_tile(qp + (size_t)rowbase * DMODEL + kt * 32, DMODEL);

  float mr[8], li[8];
#pragma unroll
  for (int i = 0; i < 8; ++i) {
    mr[i] = Mrow[rowbase + rb + i];
    li[i] = 1.f / Lrow[rowbase + rb + i];
  }

  v8f acc0 = {}, acc1 = {};
  const int d0 = wave * 2, d1 = wave * 2 + 1;

  for (int jc = 0; jc < NSEQ / 128; ++jc) {
    // stage 1: S tile at columns jb..jb+15
    const int jb = jc * 128 + wave * 16;
    v8f s = {};
#pragma unroll
    for (int kt = 0; kt < 8; ++kt) {
      v16bf b = load_b_tile(kp + (size_t)jb * DMODEL + kt * 32, DMODEL);
      s = wmma_bf16(qa[kt], b, s);
    }
#pragma unroll
    for (int i = 0; i < 8; ++i) {
      const size_t idx = (size_t)(rowbase + rb + i) * NSEQ + jb + col;
      float p = __expf(s[i] * ATTN_SCALE + mask[idx] - mr[i]) * li[i] + pref[idx];
      ldsP[(rb + i) * 128 + wave * 16 + col] = (bf16)p;
    }
    __syncthreads();
    // stage 2: accumulate P[16x128] @ V chunk
#pragma unroll
    for (int kc = 0; kc < 4; ++kc) {
      v16bf a  = load_a_tile(ldsP + kc * 32, 128);
      v16bf b0 = load_b_tile(vpT + (size_t)d0 * 16 * NSEQ + jc * 128 + kc * 32, NSEQ);
      acc0 = wmma_bf16(a, b0, acc0);
      v16bf b1 = load_b_tile(vpT + (size_t)d1 * 16 * NSEQ + jc * 128 + kc * 32, NSEQ);
      acc1 = wmma_bf16(a, b1, acc1);
    }
    __syncthreads();
  }
#pragma unroll
  for (int i = 0; i < 8; ++i) {
    avb[(size_t)(rowbase + rb + i) * DMODEL + d0 * 16 + col] = (bf16)acc0[i];
    avb[(size_t)(rowbase + rb + i) * DMODEL + d1 * 16 + col] = (bf16)acc1[i];
  }
}

// ---- GEMM(+bias) + residual + LayerNorm, Nout fixed to 256 -----------------
// Workgroup: 16 rows; wave w owns output column tiles 2w, 2w+1.
__global__ void gemm_res_ln_kernel(const bf16* __restrict__ A, int K,
                                   const bf16* __restrict__ Bt,
                                   const float* __restrict__ bias,
                                   const float* __restrict__ resid,
                                   const float* __restrict__ gamma,
                                   const float* __restrict__ beta,
                                   float* __restrict__ out_f32,
                                   bf16* __restrict__ out_bf16) {
  __shared__ float xrow[16 * 256];
  __shared__ float mu[16], rstd[16];
  const int wave = threadIdx.x >> 5, lane = threadIdx.x & 31;
  const int rowbase = blockIdx.x * 16;
  const int col = lane & 15, rb = (lane >> 4) << 3;
  const int d0 = wave * 2, d1 = wave * 2 + 1;

  v8f acc0 = {}, acc1 = {};
  for (int kt = 0; kt < (K >> 5); ++kt) {
    v16bf a  = load_a_tile(A + (size_t)rowbase * K + kt * 32, K);
    v16bf b0 = load_b_tile(Bt + (size_t)d0 * 16 * K + kt * 32, K);
    acc0 = wmma_bf16(a, b0, acc0);
    v16bf b1 = load_b_tile(Bt + (size_t)d1 * 16 * K + kt * 32, K);
    acc1 = wmma_bf16(a, b1, acc1);
  }
  const float bv0 = bias[d0 * 16 + col], bv1 = bias[d1 * 16 + col];
#pragma unroll
  for (int i = 0; i < 8; ++i) {
    const int r = rb + i;
    xrow[r * 256 + d0 * 16 + col] =
        acc0[i] + bv0 + resid[(size_t)(rowbase + r) * 256 + d0 * 16 + col];
    xrow[r * 256 + d1 * 16 + col] =
        acc1[i] + bv1 + resid[(size_t)(rowbase + r) * 256 + d1 * 16 + col];
  }
  __syncthreads();
  if (threadIdx.x < 16) {
    const int r = threadIdx.x;
    float s = 0.f;
    for (int c = 0; c < 256; ++c) s += xrow[r * 256 + c];
    const float mean = s * (1.f / 256.f);
    float v = 0.f;
    for (int c = 0; c < 256; ++c) { float d = xrow[r * 256 + c] - mean; v += d * d; }
    mu[r]   = mean;
    rstd[r] = rsqrtf(v * (1.f / 256.f) + LN_EPS);
  }
  __syncthreads();
  {
    const int r  = threadIdx.x >> 4;
    const int cb = (threadIdx.x & 15) * 16;
    const float mean = mu[r], rs = rstd[r];
    for (int j = 0; j < 16; ++j) {
      const int c = cb + j;
      const float y = (xrow[r * 256 + c] - mean) * rs * gamma[c] + beta[c];
      out_f32[(size_t)(rowbase + r) * 256 + c] = y;
      if (out_bf16) out_bf16[(size_t)(rowbase + r) * 256 + c] = (bf16)y;
    }
  }
}

// ---------------------------------------------------------------------------
extern "C" void kernel_launch(void* const* d_in, const int* in_sizes, int n_in,
                              void* d_out, int out_size, void* d_ws, size_t ws_size,
                              hipStream_t stream) {
  const float* q    = (const float*)d_in[0];
  const float* k    = (const float*)d_in[1];
  const float* v    = (const float*)d_in[2];
  const float* pref = (const float*)d_in[3];
  const float* mask = (const float*)d_in[4];
  const float* Wq = (const float*)d_in[5];  const float* bq = (const float*)d_in[6];
  const float* Wk = (const float*)d_in[7];  const float* bk = (const float*)d_in[8];
  const float* Wv = (const float*)d_in[9];  const float* bv = (const float*)d_in[10];
  const float* Wo = (const float*)d_in[11]; const float* bo = (const float*)d_in[12];
  const float* W1 = (const float*)d_in[13]; const float* b1 = (const float*)d_in[14];
  const float* W2 = (const float*)d_in[15]; const float* b2 = (const float*)d_in[16];
  const float* g1 = (const float*)d_in[17]; const float* be1 = (const float*)d_in[18];
  const float* g2 = (const float*)d_in[19]; const float* be2 = (const float*)d_in[20];

  char* w = (char*)d_ws;
  auto carve = [&](size_t bytes) -> char* {
    char* p = w;
    w += (bytes + 255) & ~(size_t)255;
    return p;
  };
  const size_t ND = (size_t)NSEQ * DMODEL;
  bf16* qb  = (bf16*)carve(ND * 2);
  bf16* kb  = (bf16*)carve(ND * 2);
  bf16* vb  = (bf16*)carve(ND * 2);
  bf16* WqT = (bf16*)carve(DMODEL * DMODEL * 2);
  bf16* WkT = (bf16*)carve(DMODEL * DMODEL * 2);
  bf16* WvT = (bf16*)carve(DMODEL * DMODEL * 2);
  bf16* WoT = (bf16*)carve(DMODEL * DMODEL * 2);
  bf16* W1T = (bf16*)carve(DMODEL * DHID * 2);
  bf16* W2T = (bf16*)carve(DHID * DMODEL * 2);
  bf16* qp  = (bf16*)carve(ND * 2);
  bf16* kp  = (bf16*)carve(ND * 2);
  bf16* vpT = (bf16*)carve(ND * 2);
  float* Mr = (float*)carve(NSEQ * 4);
  float* Lr = (float*)carve(NSEQ * 4);
  bf16* avb = (bf16*)carve(ND * 2);
  float* xf = (float*)carve(ND * 4);
  bf16* xb  = (bf16*)carve(ND * 2);
  bf16* h   = (bf16*)carve((size_t)NSEQ * DHID * 2);

  const dim3 blk(256);

  // prep: activations -> bf16, weights -> bf16 transposed [out,K]
  cvt_bf16_kernel<<<(int)(ND / 256), blk, 0, stream>>>(q, qb, (int)ND);
  cvt_bf16_kernel<<<(int)(ND / 256), blk, 0, stream>>>(k, kb, (int)ND);
  cvt_bf16_kernel<<<(int)(ND / 256), blk, 0, stream>>>(v, vb, (int)ND);
  transpose_cvt_kernel<<<256, blk, 0, stream>>>(Wq, WqT, DMODEL, DMODEL);
  transpose_cvt_kernel<<<256, blk, 0, stream>>>(Wk, WkT, DMODEL, DMODEL);
  transpose_cvt_kernel<<<256, blk, 0, stream>>>(Wv, WvT, DMODEL, DMODEL);
  transpose_cvt_kernel<<<256, blk, 0, stream>>>(Wo, WoT, DMODEL, DMODEL);
  transpose_cvt_kernel<<<1024, blk, 0, stream>>>(W1, W1T, DMODEL, DHID);
  transpose_cvt_kernel<<<1024, blk, 0, stream>>>(W2, W2T, DHID, DMODEL);

  // QKV projections: (8192/16)*(256/16)=8192 tiles, 8 tiles/block
  wmma_gemm_bias_kernel<<<1024, blk, 0, stream>>>(qb, WqT, bq, qp, NSEQ, DMODEL, DMODEL, 0, 0);
  wmma_gemm_bias_kernel<<<1024, blk, 0, stream>>>(kb, WkT, bk, kp, NSEQ, DMODEL, DMODEL, 0, 0);
  wmma_gemm_bias_kernel<<<1024, blk, 0, stream>>>(vb, WvT, bv, vpT, NSEQ, DMODEL, DMODEL, 0, 1);

  // attention: softmax stats then P@V (pref folded into P)
  softmax_stats_kernel<<<NSEQ / 128, blk, 0, stream>>>(qp, kp, mask, Mr, Lr);
  attn_pv_kernel<<<NSEQ / 16, blk, 0, stream>>>(qp, kp, vpT, mask, pref, Mr, Lr, avb);

  // out-projection + residual(q) + LN1 -> x (f32 + bf16)
  gemm_res_ln_kernel<<<NSEQ / 16, blk, 0, stream>>>(avb, DMODEL, WoT, bo, q, g1, be1, xf, xb);

  // FFN1 with ReLU: (8192/16)*(1024/16)=32768 tiles
  wmma_gemm_bias_kernel<<<4096, blk, 0, stream>>>(xb, W1T, b1, h, NSEQ, DHID, DMODEL, 1, 0);

  // FFN2 + residual(x) + LN2 -> output
  gemm_res_ln_kernel<<<NSEQ / 16, blk, 0, stream>>>(h, DHID, W2T, b2, xf, g2, be2,
                                                    (float*)d_out, nullptr);
}
